// AttentionOccupancyNetwork_4002909520789
// MI455X (gfx1250) — compile-verified, matrix operand order checked by reading, not run on hardware
//
#include <hip/hip_runtime.h>
#include <hip/hip_bf16.h>
#include <math.h>

// Problem constants (from reference)
#define BQ     16384
#define NCTX   8192
#define DMODEL 512
#define NH     8
#define DHD    64
#define KNN    16

typedef __attribute__((ext_vector_type(16))) __bf16 bf16x16;
typedef __attribute__((ext_vector_type(8)))  float  f32x8;
typedef __attribute__((ext_vector_type(4)))  unsigned int u32x4;
typedef __attribute__((ext_vector_type(8)))  int i32x8;
typedef __attribute__((ext_vector_type(4)))  int i32x4;

#if defined(__has_builtin)
#  if __has_builtin(__builtin_amdgcn_tensor_load_to_lds)
#    define HAVE_TDM 1
#  endif
#endif
#ifndef HAVE_TDM
#  define HAVE_TDM 0
#endif

__device__ __forceinline__ float gelu_erf(float x) {
  return 0.5f * x * (1.0f + erff(x * 0.70710678118654752f));
}

__device__ __forceinline__ __bf16 f2bf(float f) {
  union { float f; uint32_t u; } in; in.f = f;
  uint32_t u = in.u;
  uint32_t r = (u + 0x7fffu + ((u >> 16) & 1u)) >> 16;   // RNE
  union { unsigned short s; __bf16 h; } out; out.s = (unsigned short)r;
  return out.h;
}

// byte offset of a workgroup-shared object inside LDS
__device__ __forceinline__ unsigned lds_off_of(const void* p) {
  return (unsigned)(size_t)(const __attribute__((address_space(3))) char*)p;
}

// ---------------------------------------------------------------------------
// TDM: async 2D tile load (64 rows x 32 bf16) global -> LDS. LDS rows are
// padded by the TDM itself: 64B data + 8B pad -> 72B (18-dword) stride, which
// is 8B-aligned (enables ds b64 reads) and bank-conflict free (18 odd*2).
// ---------------------------------------------------------------------------
#if HAVE_TDM
__device__ __forceinline__ void tdm_load_tile(unsigned lds_byte_off,
                                              const __bf16* gbase,
                                              unsigned row0, unsigned k0,
                                              unsigned rows_total, unsigned Kd) {
  const unsigned long long ga =
      (unsigned long long)(size_t)gbase + ((unsigned long long)row0 * Kd + k0) * 2ull;
  u32x4 g0;
  g0[0] = 1u;                                               // count=1 (valid D#)
  g0[1] = lds_byte_off;                                     // lds_addr
  g0[2] = (unsigned)ga;                                     // global_addr lo
  g0[3] = (unsigned)((ga >> 32) & 0x01FFFFFFull) | (2u << 30); // ga hi | type=2
  i32x8 g1;
  g1[0] = (1 << 16)            // data_size = 2 bytes
        | (1 << 20)            // pad_enable
        | (3 << 22)            // pad_interval = 16 dwords (64B)
        | (1 << 25);           // pad_amount   = 2 dwords (8B)
  g1[1] = (int)((Kd & 0xFFFFu) << 16);                        // tensor_dim0 lo16
  g1[2] = (int)((Kd >> 16) | ((rows_total & 0xFFFFu) << 16)); // dim0 hi | dim1 lo
  g1[3] = (int)((rows_total >> 16) | (32u << 16));            // dim1 hi | tile_dim0=32
  g1[4] = 64;                                                 // tile_dim1=64, tile_dim2=0
  g1[5] = (int)Kd;                                            // tensor_dim0_stride lo
  g1[6] = 0;
  g1[7] = 0;
  i32x4 z4 = {0, 0, 0, 0};
#if defined(__clang_major__) && (__clang_major__ >= 23)
  i32x8 z8 = {0, 0, 0, 0, 0, 0, 0, 0};
  __builtin_amdgcn_tensor_load_to_lds(g0, g1, z4, z4, z8, 0);
#else
  __builtin_amdgcn_tensor_load_to_lds(g0, g1, z4, z4, 0);
#endif
}
#endif

// ---------------------------------------------------------------------------
// bf16 WMMA GEMM:  Y[M,N] = act( A[M,Kd] @ Wt[N,Kd]^T + bias[N] (+residual) )
// A, Wt are bf16, K-major. block = 128 threads (4 waves), tile 64x64, BK=32.
// TDM double-buffered staging: wave0 issues 2 tensor_load_to_lds for the next
// k-step while all waves run WMMA on the current buffers (s_wait_tensorcnt).
// ---------------------------------------------------------------------------
#define LDSS    36           // bf16 elems per LDS row (64B data + 8B pad)
#define TILEE   (64 * LDSS)  // elems per staged 64x32 tile

// 16x32 bf16 fragment per CDNA5 16-bit A layout (B uses same layout on Wt rows)
__device__ __forceinline__ bf16x16 load_frag(const __bf16* __restrict__ s, int lane) {
  const int row  = lane & 15;
  const int koff = (lane >> 4) * 8;
  const uint32_t* s32 = (const uint32_t*)(s + row * LDSS);
  bf16x16 f;
#pragma unroll
  for (int j = 0; j < 8; ++j) {
    const int kk = koff + ((j & 3) * 2) + ((j >> 2) * 16);
    union { uint32_t u; __bf16 h[2]; } cv; cv.u = s32[kk >> 1];
    f[2 * j]     = cv.h[0];
    f[2 * j + 1] = cv.h[1];
  }
  return f;
}

__global__ void __launch_bounds__(128)
gemm_bf16_wmma(const __bf16* __restrict__ A, const __bf16* __restrict__ Wt,
               const float* __restrict__ bias, const float* __restrict__ residual,
               float* __restrict__ Yf, __bf16* __restrict__ Ybf,
               int M, int N, int Kd, int act)
{
  // [buf][As|Bs] : 4 tiles of 64x36 bf16 = 18432 B total
  __shared__ __align__(16) __bf16 smem[4 * TILEE];

  const int tid  = threadIdx.x;
  const int wave = tid >> 5;
  const int lane = tid & 31;
  const int m0   = blockIdx.x * 64;
  const int n0   = blockIdx.y * 64;

  f32x8 acc[4];
#pragma unroll
  for (int t = 0; t < 4; ++t)
#pragma unroll
    for (int r = 0; r < 8; ++r) acc[t][r] = 0.0f;

#if HAVE_TDM
  const bool issuer = (wave == 0);           // uniform per wave; TDM ignores EXEC
  const unsigned lds0 = lds_off_of(smem);
  if (issuer) {
    tdm_load_tile(lds0 + 0 * TILEE * 2, A,  (unsigned)m0, 0u, (unsigned)M, (unsigned)Kd);
    tdm_load_tile(lds0 + 1 * TILEE * 2, Wt, (unsigned)n0, 0u, (unsigned)N, (unsigned)Kd);
    __builtin_amdgcn_s_wait_tensorcnt((short)0);
  }
  __syncthreads();

  int cur = 0;
  for (int k0 = 0; k0 < Kd; k0 += 32) {
    if (issuer && (k0 + 32 < Kd)) {          // prefetch next k-step into other buffer
      const unsigned nb = lds0 + (unsigned)(cur ^ 1) * 2u * TILEE * 2u;
      tdm_load_tile(nb + 0 * TILEE * 2, A,  (unsigned)m0, (unsigned)(k0 + 32),
                    (unsigned)M, (unsigned)Kd);
      tdm_load_tile(nb + 1 * TILEE * 2, Wt, (unsigned)n0, (unsigned)(k0 + 32),
                    (unsigned)N, (unsigned)Kd);
    }
    const __bf16* As = smem + cur * 2 * TILEE;
    const __bf16* Bs = As + TILEE;
    const bf16x16 a = load_frag(As + (wave * 16) * LDSS, lane);
#pragma unroll
    for (int t = 0; t < 4; ++t) {
      const bf16x16 b = load_frag(Bs + (t * 16) * LDSS, lane);
      acc[t] = __builtin_amdgcn_wmma_f32_16x16x32_bf16(
          false, a, false, b, (short)0, acc[t], false, false);
    }
    if (issuer) __builtin_amdgcn_s_wait_tensorcnt((short)0);
    __syncthreads();
    cur ^= 1;
  }
#else
  // fallback: manual bf16 staging, single buffer
  __bf16* As = smem;
  __bf16* Bs = smem + TILEE;
  const int lrow = tid >> 1;   // 0..63
  const int half = tid & 1;
  for (int k0 = 0; k0 < Kd; k0 += 32) {
    {
      const uint32_t* src = (const uint32_t*)(A + (size_t)(m0 + lrow) * Kd + k0) + half * 8;
      uint32_t* dst = (uint32_t*)(As + lrow * LDSS) + half * 8;
#pragma unroll
      for (int j = 0; j < 8; ++j) dst[j] = src[j];
    }
    {
      const uint32_t* src = (const uint32_t*)(Wt + (size_t)(n0 + lrow) * Kd + k0) + half * 8;
      uint32_t* dst = (uint32_t*)(Bs + lrow * LDSS) + half * 8;
#pragma unroll
      for (int j = 0; j < 8; ++j) dst[j] = src[j];
    }
    __syncthreads();
    const bf16x16 a = load_frag(As + (wave * 16) * LDSS, lane);
#pragma unroll
    for (int t = 0; t < 4; ++t) {
      const bf16x16 b = load_frag(Bs + (t * 16) * LDSS, lane);
      acc[t] = __builtin_amdgcn_wmma_f32_16x16x32_bf16(
          false, a, false, b, (short)0, acc[t], false, false);
    }
    __syncthreads();
  }
#endif

  // epilogue: C/D layout -> M = r + (lane>=16)*8, N = lane&15
  const int mr    = lane & 15;
  const int rbase = (lane >> 4) * 8;
  float v[4][8];
#pragma unroll
  for (int t = 0; t < 4; ++t)
#pragma unroll
    for (int r = 0; r < 8; ++r)
      v[t][r] = acc[t][r] + bias[n0 + t * 16 + mr];
  if (residual) {
#pragma unroll
    for (int t = 0; t < 4; ++t)
#pragma unroll
      for (int r = 0; r < 8; ++r)
        v[t][r] += residual[(size_t)(m0 + wave * 16 + rbase + r) * N + (n0 + t * 16 + mr)];
  }
  if (act) {
#pragma unroll
    for (int t = 0; t < 4; ++t)
#pragma unroll
      for (int r = 0; r < 8; ++r) v[t][r] = gelu_erf(v[t][r]);
  }
  if (Yf) {
#pragma unroll
    for (int t = 0; t < 4; ++t)
#pragma unroll
      for (int r = 0; r < 8; ++r)
        Yf[(size_t)(m0 + wave * 16 + rbase + r) * N + (n0 + t * 16 + mr)] = v[t][r];
  }
  if (Ybf) {
#pragma unroll
    for (int t = 0; t < 4; ++t)
#pragma unroll
      for (int r = 0; r < 8; ++r)
        Ybf[(size_t)(m0 + wave * 16 + rbase + r) * N + (n0 + t * 16 + mr)] = f2bf(v[t][r]);
  }
}

// ---------------------------------------------------------------------------
// fp32 -> bf16 elementwise convert (one-time per operand)
// ---------------------------------------------------------------------------
__global__ void convert_bf16_kernel(const float* __restrict__ src,
                                    __bf16* __restrict__ dst, int n) {
  const int t = blockIdx.x * blockDim.x + threadIdx.x;
  if (t < n) dst[t] = f2bf(src[t]);
}

// W[Kd][N] fp32 -> Wt[N][Kd] bf16 (K-major for WMMA B operand)
__global__ void transpose_w_kernel(const float* __restrict__ W,
                                   __bf16* __restrict__ Wt, int Kd, int N) {
  const int t = blockIdx.x * blockDim.x + threadIdx.x;
  if (t >= Kd * N) return;
  const int n = t / Kd, k = t - n * Kd;
  Wt[t] = f2bf(W[(size_t)k * N + n]);
}

// ---------------------------------------------------------------------------
// Fold biases / rel-projection:  bK = bk + brel@Wk ; WrelK = Wrel@Wk  (same V)
// ---------------------------------------------------------------------------
__global__ void consts_kernel(const float* __restrict__ brel, const float* __restrict__ Wrel,
                              const float* __restrict__ Wk,   const float* __restrict__ bk,
                              const float* __restrict__ Wv,   const float* __restrict__ bv,
                              float* __restrict__ bK, float* __restrict__ bV,
                              float* __restrict__ WrelK, float* __restrict__ WrelV)
{
  const int n = blockIdx.x * blockDim.x + threadIdx.x;
  if (n >= DMODEL) return;
  float sk = bk[n], sv = bv[n];
  float rk0 = 0.f, rk1 = 0.f, rk2 = 0.f, rv0 = 0.f, rv1 = 0.f, rv2 = 0.f;
  for (int d = 0; d < DMODEL; ++d) {
    const float wkv = Wk[(size_t)d * DMODEL + n];
    const float wvv = Wv[(size_t)d * DMODEL + n];
    const float br  = brel[d];
    sk += br * wkv;                    sv += br * wvv;
    rk0 += Wrel[d] * wkv;              rv0 += Wrel[d] * wvv;
    rk1 += Wrel[DMODEL + d] * wkv;     rv1 += Wrel[DMODEL + d] * wvv;
    rk2 += Wrel[2 * DMODEL + d] * wkv; rv2 += Wrel[2 * DMODEL + d] * wvv;
  }
  bK[n] = sk; bV[n] = sv;
  WrelK[n] = rk0; WrelK[DMODEL + n] = rk1; WrelK[2 * DMODEL + n] = rk2;
  WrelV[n] = rv0; WrelV[DMODEL + n] = rv1; WrelV[2 * DMODEL + n] = rv2;
}

// ---------------------------------------------------------------------------
// Query encoder layer 1 -> bf16 hidden (3 -> 512)
// ---------------------------------------------------------------------------
__global__ void qe_hidden_kernel(const float* __restrict__ qc, const float* __restrict__ W1,
                                 const float* __restrict__ b1, __bf16* __restrict__ hid)
{
  const int t = blockIdx.x * blockDim.x + threadIdx.x;
  if (t >= BQ * DMODEL) return;
  const int b = t >> 9, n = t & (DMODEL - 1);
  const float x = qc[b * 3 + 0] * W1[n] + qc[b * 3 + 1] * W1[DMODEL + n]
                + qc[b * 3 + 2] * W1[2 * DMODEL + n] + b1[n];
  hid[t] = f2bf(gelu_erf(x));
}

// ---------------------------------------------------------------------------
// Top-16 nearest context points; one wave per query (wave32 butterfly merge).
// ---------------------------------------------------------------------------
__global__ void __launch_bounds__(128)
topk_kernel(const float* __restrict__ qc, const float* __restrict__ cc, int* __restrict__ nn)
{
  const int wave = threadIdx.x >> 5, lane = threadIdx.x & 31;
  const int b = blockIdx.x * 4 + wave;
  const float qx = qc[b * 3 + 0], qy = qc[b * 3 + 1], qz = qc[b * 3 + 2];

  float ad[KNN]; int ai[KNN];
#pragma unroll
  for (int j = 0; j < KNN; ++j) { ad[j] = 3.4e38f; ai[j] = 0; }

  for (int c = lane; c < NCTX; c += 32) {
    const float dx = qx - cc[c * 3 + 0];
    const float dy = qy - cc[c * 3 + 1];
    const float dz = qz - cc[c * 3 + 2];
    float d = dx * dx + dy * dy + dz * dz;
    if (d < ad[KNN - 1]) {
      float cd = d; int ci = c;
#pragma unroll
      for (int j = 0; j < KNN; ++j) {
        if (cd < ad[j]) {
          float td = ad[j]; ad[j] = cd; cd = td;
          int   ti = ai[j]; ai[j] = ci; ci = ti;
        }
      }
    }
  }

  int p = 0;
  for (int r = 0; r < KNN; ++r) {
    float hd = ad[0]; int hi = ai[0];
#pragma unroll
    for (int j = 1; j < KNN; ++j) if (p == j) { hd = ad[j]; hi = ai[j]; }
    if (p >= KNN) hd = 3.4e38f;
    float bd = hd; int bi = hi; int bo = lane;
#pragma unroll
    for (int s = 0; s < 5; ++s) {
      const int off = 16 >> s;
      const float od = __shfl_xor(bd, off, 32);
      const int   oi = __shfl_xor(bi, off, 32);
      const int   oo = __shfl_xor(bo, off, 32);
      if (od < bd || (od == bd && oo < bo)) { bd = od; bi = oi; bo = oo; }
    }
    if (lane == 0) nn[b * KNN + r] = bi;
    if (lane == bo) p++;
  }
}

// ---------------------------------------------------------------------------
// Cross attention; one wave per query. Head h = lanes 4h..4h+3 (16 f32 each).
// CK/CV (32MB) are L2-resident on MI455X (192MB L2). Output: bf16 agg.
// ---------------------------------------------------------------------------
__global__ void __launch_bounds__(128)
attn_kernel(const float* __restrict__ qvec, const int* __restrict__ nn,
            const float* __restrict__ CK, const float* __restrict__ CV,
            const float* __restrict__ qc, const float* __restrict__ cc,
            const float* __restrict__ WrelK, const float* __restrict__ WrelV,
            __bf16* __restrict__ agg)
{
  const int wave = threadIdx.x >> 5, lane = threadIdx.x & 31;
  const int b  = blockIdx.x * 4 + wave;
  const int c0 = lane * 16;

  float ql[16], wk0[16], wk1[16], wk2[16], wv0[16], wv1[16], wv2[16];
#pragma unroll
  for (int j = 0; j < 4; ++j) {
    float4 v;
    v = *(const float4*)(qvec + (size_t)b * DMODEL + c0 + 4 * j);
    ql[4*j] = v.x; ql[4*j+1] = v.y; ql[4*j+2] = v.z; ql[4*j+3] = v.w;
    v = *(const float4*)(WrelK + c0 + 4 * j);
    wk0[4*j] = v.x; wk0[4*j+1] = v.y; wk0[4*j+2] = v.z; wk0[4*j+3] = v.w;
    v = *(const float4*)(WrelK + DMODEL + c0 + 4 * j);
    wk1[4*j] = v.x; wk1[4*j+1] = v.y; wk1[4*j+2] = v.z; wk1[4*j+3] = v.w;
    v = *(const float4*)(WrelK + 2 * DMODEL + c0 + 4 * j);
    wk2[4*j] = v.x; wk2[4*j+1] = v.y; wk2[4*j+2] = v.z; wk2[4*j+3] = v.w;
    v = *(const float4*)(WrelV + c0 + 4 * j);
    wv0[4*j] = v.x; wv0[4*j+1] = v.y; wv0[4*j+2] = v.z; wv0[4*j+3] = v.w;
    v = *(const float4*)(WrelV + DMODEL + c0 + 4 * j);
    wv1[4*j] = v.x; wv1[4*j+1] = v.y; wv1[4*j+2] = v.z; wv1[4*j+3] = v.w;
    v = *(const float4*)(WrelV + 2 * DMODEL + c0 + 4 * j);
    wv2[4*j] = v.x; wv2[4*j+1] = v.y; wv2[4*j+2] = v.z; wv2[4*j+3] = v.w;
  }
  const float qx = qc[b * 3 + 0], qy = qc[b * 3 + 1], qz = qc[b * 3 + 2];

  float s[KNN];
#pragma unroll
  for (int k = 0; k < KNN; ++k) {
    const int id = nn[b * KNN + k];
    const float rx = qx - cc[id * 3 + 0];
    const float ry = qy - cc[id * 3 + 1];
    const float rz = qz - cc[id * 3 + 2];
    const float* kr = CK + (size_t)id * DMODEL + c0;
    float part = 0.f;
#pragma unroll
    for (int j = 0; j < 4; ++j) {
      float4 v = *(const float4*)(kr + 4 * j);
      part += ql[4*j+0] * (v.x + rx * wk0[4*j+0] + ry * wk1[4*j+0] + rz * wk2[4*j+0]);
      part += ql[4*j+1] * (v.y + rx * wk0[4*j+1] + ry * wk1[4*j+1] + rz * wk2[4*j+1]);
      part += ql[4*j+2] * (v.z + rx * wk0[4*j+2] + ry * wk1[4*j+2] + rz * wk2[4*j+2]);
      part += ql[4*j+3] * (v.w + rx * wk0[4*j+3] + ry * wk1[4*j+3] + rz * wk2[4*j+3]);
    }
    part += __shfl_xor(part, 1, 32);
    part += __shfl_xor(part, 2, 32);
    s[k] = part * 0.125f;        // 1/sqrt(64)
  }
  float m = s[0];
#pragma unroll
  for (int k = 1; k < KNN; ++k) m = fmaxf(m, s[k]);
  float sum = 0.f;
#pragma unroll
  for (int k = 0; k < KNN; ++k) { s[k] = expf(s[k] - m); sum += s[k]; }
  const float inv = 1.0f / sum;
#pragma unroll
  for (int k = 0; k < KNN; ++k) s[k] *= inv;

  float out[16];
#pragma unroll
  for (int i = 0; i < 16; ++i) out[i] = 0.f;
#pragma unroll
  for (int k = 0; k < KNN; ++k) {
    const int id = nn[b * KNN + k];
    const float rx = qx - cc[id * 3 + 0];
    const float ry = qy - cc[id * 3 + 1];
    const float rz = qz - cc[id * 3 + 2];
    const float* vr = CV + (size_t)id * DMODEL + c0;
    const float ak = s[k];
#pragma unroll
    for (int j = 0; j < 4; ++j) {
      float4 v = *(const float4*)(vr + 4 * j);
      out[4*j+0] += ak * (v.x + rx * wv0[4*j+0] + ry * wv1[4*j+0] + rz * wv2[4*j+0]);
      out[4*j+1] += ak * (v.y + rx * wv0[4*j+1] + ry * wv1[4*j+1] + rz * wv2[4*j+1]);
      out[4*j+2] += ak * (v.z + rx * wv0[4*j+2] + ry * wv1[4*j+2] + rz * wv2[4*j+2]);
      out[4*j+3] += ak * (v.w + rx * wv0[4*j+3] + ry * wv1[4*j+3] + rz * wv2[4*j+3]);
    }
  }
#pragma unroll
  for (int i = 0; i < 16; ++i)
    agg[(size_t)b * DMODEL + c0 + i] = f2bf(out[i]);
}

// ---------------------------------------------------------------------------
// LayerNorm over D=512: read f32 x, write bf16 for the head GEMM.
// ---------------------------------------------------------------------------
__global__ void __launch_bounds__(128)
layernorm_kernel(const float* __restrict__ x, const float* __restrict__ g,
                 const float* __restrict__ bta, __bf16* __restrict__ xb)
{
  const int wave = threadIdx.x >> 5, lane = threadIdx.x & 31;
  const int row = blockIdx.x * 4 + wave;
  float v[16];
  float s = 0.f, s2 = 0.f;
#pragma unroll
  for (int j = 0; j < 4; ++j) {
    float4 t = *(const float4*)(x + (size_t)row * DMODEL + lane * 16 + 4 * j);
    v[4*j] = t.x; v[4*j+1] = t.y; v[4*j+2] = t.z; v[4*j+3] = t.w;
  }
#pragma unroll
  for (int i = 0; i < 16; ++i) { s += v[i]; s2 += v[i] * v[i]; }
#pragma unroll
  for (int o = 0; o < 5; ++o) {
    s  += __shfl_xor(s,  16 >> o, 32);
    s2 += __shfl_xor(s2, 16 >> o, 32);
  }
  const float mu  = s * (1.0f / DMODEL);
  const float var = s2 * (1.0f / DMODEL) - mu * mu;
  const float rin = rsqrtf(var + 1e-5f);
#pragma unroll
  for (int i = 0; i < 16; ++i) {
    const int c = lane * 16 + i;
    xb[(size_t)row * DMODEL + c] = f2bf((v[i] - mu) * rin * g[c] + bta[c]);
  }
}

// ---------------------------------------------------------------------------
// logits = h2[B,256] @ Wh3[256,1] + bh3 ; one wave per row.
// ---------------------------------------------------------------------------
__global__ void __launch_bounds__(128)
logits_kernel(const float* __restrict__ h2, const float* __restrict__ Wh3,
              const float* __restrict__ bh3, float* __restrict__ out)
{
  const int wave = threadIdx.x >> 5, lane = threadIdx.x & 31;
  const int row = blockIdx.x * 4 + wave;
  float part = 0.f;
#pragma unroll
  for (int i = 0; i < 8; ++i) {
    const int c = lane * 8 + i;
    part += h2[(size_t)row * 256 + c] * Wh3[c];
  }
#pragma unroll
  for (int o = 0; o < 5; ++o) part += __shfl_xor(part, 16 >> o, 32);
  if (lane == 0) out[row] = part + bh3[0];
}

// ---------------------------------------------------------------------------
extern "C" void kernel_launch(void* const* d_in, const int* in_sizes, int n_in,
                              void* d_out, int out_size, void* d_ws, size_t ws_size,
                              hipStream_t stream) {
  const float* qc   = (const float*)d_in[0];
  const float* ccd  = (const float*)d_in[1];
  const float* cf   = (const float*)d_in[2];
  const float* Wqe1 = (const float*)d_in[3];
  const float* bqe1 = (const float*)d_in[4];
  const float* Wqe2 = (const float*)d_in[5];
  const float* bqe2 = (const float*)d_in[6];
  const float* Wrel = (const float*)d_in[7];
  const float* brel = (const float*)d_in[8];
  const float* Wq   = (const float*)d_in[9];
  const float* bq   = (const float*)d_in[10];
  const float* Wk   = (const float*)d_in[11];
  const float* bk   = (const float*)d_in[12];
  const float* Wv   = (const float*)d_in[13];
  const float* bv   = (const float*)d_in[14];
  const float* Wo   = (const float*)d_in[15];
  const float* bo   = (const float*)d_in[16];
  const float* ln_g = (const float*)d_in[17];
  const float* ln_b = (const float*)d_in[18];
  const float* Wh1  = (const float*)d_in[19];
  const float* bh1  = (const float*)d_in[20];
  const float* Wh2  = (const float*)d_in[21];
  const float* bh2  = (const float*)d_in[22];
  const float* Wh3  = (const float*)d_in[23];
  const float* bh3  = (const float*)d_in[24];
  float* out = (float*)d_out;

  // ---- workspace layout (bytes) ----
  char* ws = (char*)d_ws;
  const size_t MB = 1024ull * 1024ull;
  float*  CK    = (float*)(ws + 0);          // 16MB   (later: x f32, 32MB @0)
  float*  CV    = (float*)(ws + 16 * MB);    // 16MB
  float*  xbuf  = CK;
  float*  qe    = (float*)(ws + 32 * MB);    // 32MB f32
  float*  qvec  = (float*)(ws + 64 * MB);    // 32MB f32 (later: h2 f32 16MB)
  float*  h2    = qvec;
  __bf16* cfb   = (__bf16*)(ws + 96 * MB);   // 8MB
  __bf16* qeb   = (__bf16*)(ws + 104 * MB);  // 16MB (later: h1b)
  __bf16* h1b   = qeb;
  __bf16* hidb  = (__bf16*)(ws + 120 * MB);  // 16MB (later: aggb)
  __bf16* aggb  = hidb;
  __bf16* xb    = (__bf16*)(ws + 136 * MB);  // 16MB
  __bf16* WtK   = (__bf16*)(ws + 152 * MB);          // 512KB each
  __bf16* WtV   = (__bf16*)(ws + 152 * MB + 512 * 1024);
  __bf16* WtQe2 = (__bf16*)(ws + 153 * MB);
  __bf16* WtQ   = (__bf16*)(ws + 153 * MB + 512 * 1024);
  __bf16* WtO   = (__bf16*)(ws + 154 * MB);
  __bf16* WtH1  = (__bf16*)(ws + 154 * MB + 512 * 1024);
  __bf16* WtH2  = (__bf16*)(ws + 155 * MB);          // 256KB
  int*    nnidx = (int*)(ws + 156 * MB);             // 1MB
  float*  consts= (float*)(ws + 157 * MB);
  float*  bK    = consts;
  float*  bV    = consts + DMODEL;
  float*  WrelK = consts + 2 * DMODEL;
  float*  WrelV = consts + 5 * DMODEL;

  const int WSZ = DMODEL * DMODEL;

  // 1. fold biases / rel projections; one-time bf16 conversions + transposes
  consts_kernel<<<2, 256, 0, stream>>>(brel, Wrel, Wk, bk, Wv, bv, bK, bV, WrelK, WrelV);
  convert_bf16_kernel<<<(NCTX * DMODEL) / 256, 256, 0, stream>>>(cf, cfb, NCTX * DMODEL);
  transpose_w_kernel<<<WSZ / 256, 256, 0, stream>>>(Wk,   WtK,   DMODEL, DMODEL);
  transpose_w_kernel<<<WSZ / 256, 256, 0, stream>>>(Wv,   WtV,   DMODEL, DMODEL);
  transpose_w_kernel<<<WSZ / 256, 256, 0, stream>>>(Wqe2, WtQe2, DMODEL, DMODEL);
  transpose_w_kernel<<<WSZ / 256, 256, 0, stream>>>(Wq,   WtQ,   DMODEL, DMODEL);
  transpose_w_kernel<<<WSZ / 256, 256, 0, stream>>>(Wo,   WtO,   DMODEL, DMODEL);
  transpose_w_kernel<<<WSZ / 256, 256, 0, stream>>>(Wh1,  WtH1,  DMODEL, DMODEL);
  transpose_w_kernel<<<(DMODEL * 256) / 256, 256, 0, stream>>>(Wh2, WtH2, DMODEL, 256);
  // 2. context K/V tables (WMMA + TDM), folded biases
  gemm_bf16_wmma<<<dim3(NCTX / 64, DMODEL / 64), 128, 0, stream>>>(
      cfb, WtK, bK, nullptr, CK, nullptr, NCTX, DMODEL, DMODEL, 0);
  gemm_bf16_wmma<<<dim3(NCTX / 64, DMODEL / 64), 128, 0, stream>>>(
      cfb, WtV, bV, nullptr, CV, nullptr, NCTX, DMODEL, DMODEL, 0);
  // 3. query encoder
  qe_hidden_kernel<<<(BQ * DMODEL) / 256, 256, 0, stream>>>(qc, Wqe1, bqe1, hidb);
  gemm_bf16_wmma<<<dim3(BQ / 64, DMODEL / 64), 128, 0, stream>>>(
      hidb, WtQe2, bqe2, nullptr, qe, qeb, BQ, DMODEL, DMODEL, 0);
  // 4. Q projection
  gemm_bf16_wmma<<<dim3(BQ / 64, DMODEL / 64), 128, 0, stream>>>(
      qeb, WtQ, bq, nullptr, qvec, nullptr, BQ, DMODEL, DMODEL, 0);
  // 5. top-16 neighbors
  topk_kernel<<<BQ / 4, 128, 0, stream>>>(qc, ccd, nnidx);
  // 6. attention (CK/CV live in L2) -> bf16 agg (reuses hidb)
  attn_kernel<<<BQ / 4, 128, 0, stream>>>(qvec, nnidx, CK, CV, qc, ccd, WrelK, WrelV, aggb);
  // 7. output projection + residual(qe f32) -> x f32 (recycles CK/CV region)
  gemm_bf16_wmma<<<dim3(BQ / 64, DMODEL / 64), 128, 0, stream>>>(
      aggb, WtO, bo, qe, xbuf, nullptr, BQ, DMODEL, DMODEL, 0);
  // 8. LayerNorm -> bf16
  layernorm_kernel<<<BQ / 4, 128, 0, stream>>>(xbuf, ln_g, ln_b, xb);
  // 9. head: D->D (gelu, bf16 out), D->D/2 (gelu, f32 out), D/2->1
  gemm_bf16_wmma<<<dim3(BQ / 64, DMODEL / 64), 128, 0, stream>>>(
      xb, WtH1, bh1, nullptr, nullptr, h1b, BQ, DMODEL, DMODEL, 1);
  gemm_bf16_wmma<<<dim3(BQ / 64, 256 / 64), 128, 0, stream>>>(
      h1b, WtH2, bh2, nullptr, h2, nullptr, BQ, 256, DMODEL, 1);
  logits_kernel<<<BQ / 4, 128, 0, stream>>>(h2, Wh3, bh3, out);
  (void)in_sizes; (void)n_in; (void)out_size; (void)ws_size;
}
